// KANLayer_738734375292
// MI455X (gfx1250) — matrix operand order repaired and checked by reading, not compiled, not run on hardware
//
#include <hip/hip_runtime.h>

// ---------------------------------------------------------------------------
// KAN layer fused kernel for gfx1250 (MI455X), wave32 + WMMA bf16.
//   out = basis(x) @ spline_weight + silu(x) @ base_w.T + base_b
// Unified GEMM over K = 4096 (spline) + 512 (base) = 4608.
//  - prep kernel packs B = [spline_weight ; base_w.T] to bf16, tile-contiguous
//  - main kernel: BK=64, double-buffered LDS, async global->LDS B staging,
//    A tile synthesized on the fly (exp / silu), 16x16x32 bf16 WMMA, f32 acc.
// ---------------------------------------------------------------------------

typedef __bf16 v16bf __attribute__((ext_vector_type(16)));
typedef float  v8f   __attribute__((ext_vector_type(8)));
typedef int    v4i   __attribute__((vector_size(16)));   // matches async builtin param

typedef __attribute__((address_space(1))) v4i* glob_v4i_ptr;
typedef __attribute__((address_space(3))) v4i* lds_v4i_ptr;

#define N_ROWS   16384
#define IN_DIM   512
#define OUT_DIM  512
#define K_SPLINE 4096
#define K_TOTAL  4608
#define BK       64
#define NT       (K_TOTAL / BK)   // 72 k-tiles (spline: kt 0..63, base: 64..71)
#define BM       128
#define BN       128
#define LDSS     72               // 64 + 8 halfword pad -> 144B rows, conflict-free

#if __has_builtin(__builtin_amdgcn_global_load_async_to_lds_b128) && \
    __has_builtin(__builtin_amdgcn_s_wait_asynccnt)
#define USE_ASYNC 1
#else
#define USE_ASYNC 0
#endif

// f32x2 -> packed bf16x2, round-half-up, single v_perm_b32 for the merge.
__device__ __forceinline__ unsigned int pack2(float a, float b) {
  const unsigned int ua = __float_as_uint(a) + 0x8000u;
  const unsigned int ub = __float_as_uint(b) + 0x8000u;
  return __builtin_amdgcn_perm(ub, ua, 0x07060302u);  // {b.hi16, a.hi16}
}

__device__ __forceinline__ float silu_f(float v) {
  return v / (1.0f + __expf(-v));
}

// ---------------------------------------------------------------------------
// Prep: pack unified B matrix (K_TOTAL x OUT_DIM) to bf16, tiled so that
// tile (kt, n) is contiguous:  Bbf[((kt*OUT_DIM)+n)*BK + kk]
// B[k][n] = k < 4096 ? spline_weight[k][n] : base_w[n][k-4096]
// ---------------------------------------------------------------------------
__global__ __launch_bounds__(256) void kan_prep_weights(
    const float* __restrict__ sw, const float* __restrict__ bw,
    unsigned short* __restrict__ Bbf) {
  const int t = blockIdx.x * 256 + threadIdx.x;   // one (kt, n) pair per thread
  if (t >= NT * OUT_DIM) return;
  const int n  = t & (OUT_DIM - 1);
  const int kt = t >> 9;
  const int k0 = kt * BK;                          // tile is entirely spline or base
  unsigned int pk[32];
  if (k0 < K_SPLINE) {
#pragma unroll
    for (int j = 0; j < BK; j += 2) {              // lanes vary n -> coalesced
      const float v0 = sw[(size_t)(k0 + j)     * OUT_DIM + n];
      const float v1 = sw[(size_t)(k0 + j + 1) * OUT_DIM + n];
      pk[j >> 1] = pack2(v0, v1);
    }
  } else {
    const float4* wr4 = (const float4*)&bw[(size_t)n * IN_DIM + (k0 - K_SPLINE)];
#pragma unroll
    for (int q = 0; q < 16; ++q) {
      const float4 v = wr4[q];
      pk[q * 2 + 0] = pack2(v.x, v.y);
      pk[q * 2 + 1] = pack2(v.z, v.w);
    }
  }
  uint4* dst = (uint4*)(Bbf + (size_t)t * BK);
#pragma unroll
  for (int q = 0; q < 8; ++q)
    dst[q] = make_uint4(pk[q * 4], pk[q * 4 + 1], pk[q * 4 + 2], pk[q * 4 + 3]);
}

// ---------------------------------------------------------------------------
// Main fused GEMM kernel.  256 threads = 8 waves; block tile 128x128;
// wave tile 32x64 (2x4 frags); BK=64 -> 16 WMMAs between barriers.
// ---------------------------------------------------------------------------
__global__ __launch_bounds__(256) void kan_wmma_bf16_kernel(
    const float* __restrict__ x,                  // (N, IN_DIM) fp32
    const unsigned short* __restrict__ Bbf,       // packed bf16 B tiles
    const float* __restrict__ bb,                 // (OUT_DIM,)
    float* __restrict__ out)                      // (N, OUT_DIM) fp32
{
  __shared__ __align__(16) unsigned short Ash[2][BM * LDSS];
  __shared__ __align__(16) unsigned short Bsh[2][BN * LDSS];

  const int tid    = threadIdx.x;
  const int lane   = tid & 31;
  const int wid    = tid >> 5;
  const int wave_m = wid >> 1;   // 0..3
  const int wave_n = wid & 1;    // 0..1

  const int tile_n = blockIdx.x & 3;
  const int tile_m = blockIdx.x >> 2;
  const int r0 = tile_m * BM;
  const int n0 = tile_n * BN;

  const float h    = 4.0f / 7.0f;
  const float invh = 1.75f;

  // Every thread stages half an A row (32 k) and 64 bytes of B (async).
  const int srow  = tid >> 1;          // 0..127
  const int shalf = tid & 1;           // 0 or 1 -> k offset 0 / 32

  auto stage = [&](int kt2, int buf) {
    const int k0 = kt2 * BK;
    const int koff = shalf << 5;       // 0 or 32
    // ---- A: synthesize 32 bf16 values for row `srow`, k in [koff, koff+32) ----
    unsigned int pk[16];
    if (k0 < K_SPLINE) {
      // 4 features x 8 grid points.
      const int d0 = (k0 >> 3) + (shalf << 2);
      const float4 xv = *(const float4*)&x[(size_t)(r0 + srow) * IN_DIM + d0];
      const float xs[4] = {xv.x, xv.y, xv.z, xv.w};
#pragma unroll
      for (int dd = 0; dd < 4; ++dd) {
#pragma unroll
        for (int g = 0; g < 8; g += 2) {
          const float t0 = (xs[dd] - (-2.0f + h * (float)g))       * invh;
          const float t1 = (xs[dd] - (-2.0f + h * (float)(g + 1))) * invh;
          pk[dd * 4 + (g >> 1)] = pack2(__expf(-t0 * t0), __expf(-t1 * t1));
        }
      }
    } else {
      const float4* xr4 =
          (const float4*)&x[(size_t)(r0 + srow) * IN_DIM + (k0 - K_SPLINE) + koff];
#pragma unroll
      for (int q = 0; q < 8; ++q) {
        const float4 v = xr4[q];
        pk[q * 2 + 0] = pack2(silu_f(v.x), silu_f(v.y));
        pk[q * 2 + 1] = pack2(silu_f(v.z), silu_f(v.w));
      }
    }
    uint4* adst = (uint4*)&Ash[buf][srow * LDSS + koff];
#pragma unroll
    for (int q = 0; q < 4; ++q)
      adst[q] = make_uint4(pk[q * 4], pk[q * 4 + 1], pk[q * 4 + 2], pk[q * 4 + 3]);

    // ---- B: 64B of the contiguous 16KB pre-packed tile -> LDS ----
    const unsigned short* src =
        Bbf + ((size_t)kt2 * OUT_DIM + (n0 + srow)) * BK + koff;
#if USE_ASYNC
#pragma unroll
    for (int q = 0; q < 4; ++q) {
      __builtin_amdgcn_global_load_async_to_lds_b128(
          (glob_v4i_ptr)(src + q * 8),
          (lds_v4i_ptr)&Bsh[buf][srow * LDSS + koff + q * 8],
          0, 0);
    }
#else
    const uint4* s4 = (const uint4*)src;
    uint4* bdst = (uint4*)&Bsh[buf][srow * LDSS + koff];
#pragma unroll
    for (int q = 0; q < 4; ++q) bdst[q] = s4[q];
#endif
  };

  v8f acc[2][4] = {};

  stage(0, 0);
#if USE_ASYNC
  __builtin_amdgcn_s_wait_asynccnt(0);
#endif
  __syncthreads();

#pragma unroll 2
  for (int kt = 0; kt < NT; ++kt) {
    const int buf = kt & 1;

    // ---- fragment loads for both 32-k steps (ISA lane striping) ----
    v16bf afrag[2][2], bfrag[2][4];
    {
      const int row = lane & 15;
      const int kba = (lane < 16) ? 0 : 8;
      const int kbb = (lane < 16) ? 0 : 8;
#pragma unroll
      for (int s = 0; s < 2; ++s) {
        const int ko = s << 5;
#pragma unroll
        for (int mf = 0; mf < 2; ++mf) {
          const unsigned short* ap =
              &Ash[buf][(wave_m * 32 + mf * 16 + row) * LDSS + ko];
          union { v16bf v; uint4 q[2]; } u;
          u.q[0] = *(const uint4*)(ap + kba);
          u.q[1] = *(const uint4*)(ap + 16 + kba);
          afrag[s][mf] = u.v;
        }
#pragma unroll
        for (int nf = 0; nf < 4; ++nf) {
          const unsigned short* bp =
              &Bsh[buf][(wave_n * 64 + nf * 16 + row) * LDSS + ko];
          union { v16bf v; uint4 q[2]; } u;
          u.q[0] = *(const uint4*)(bp + kbb);
          u.q[1] = *(const uint4*)(bp + 16 + kbb);
          bfrag[s][nf] = u.v;
        }
      }
    }

    // ---- overlap: stage next tile while the 16 WMMAs run ----
    if (kt + 1 < NT) stage(kt + 1, buf ^ 1);

#pragma unroll
    for (int s = 0; s < 2; ++s)
#pragma unroll
      for (int mf = 0; mf < 2; ++mf)
#pragma unroll
        for (int nf = 0; nf < 4; ++nf)
          acc[mf][nf] = __builtin_amdgcn_wmma_f32_16x16x32_bf16(
              false, afrag[s][mf], false, bfrag[s][nf],
              (short)0, acc[mf][nf], false, false);

#if USE_ASYNC
    __builtin_amdgcn_s_wait_asynccnt(0);
#endif
    __syncthreads();
  }

  // ---- epilogue: + base_b, fp32 stores (C/D layout: VGPR v -> M=v / v+8) ----
  const int mrow_off = (lane < 16) ? 0 : 8;
  const int ncol     = lane & 15;
#pragma unroll
  for (int nf = 0; nf < 4; ++nf) {
    const int col = n0 + wave_n * 64 + nf * 16 + ncol;
    const float bbv = bb[col];
#pragma unroll
    for (int mf = 0; mf < 2; ++mf) {
      const int rbase = r0 + wave_m * 32 + mf * 16 + mrow_off;
#pragma unroll
      for (int v = 0; v < 8; ++v) {
        out[(size_t)(rbase + v) * OUT_DIM + col] = acc[mf][nf][v] + bbv;
      }
    }
  }
}

extern "C" void kernel_launch(void* const* d_in, const int* in_sizes, int n_in,
                              void* d_out, int out_size, void* d_ws, size_t ws_size,
                              hipStream_t stream) {
  (void)in_sizes; (void)n_in; (void)out_size; (void)ws_size;
  const float* x  = (const float*)d_in[0];
  // d_in[1] is the grid; fixed linspace recomputed in-kernel.
  const float* sw = (const float*)d_in[2];
  const float* bw = (const float*)d_in[3];
  const float* bb = (const float*)d_in[4];
  float* out = (float*)d_out;
  unsigned short* Bbf = (unsigned short*)d_ws;    // needs NT*OUT_DIM*BK*2 = 4.5 MB

  // Pack weights to bf16, tile-contiguous (deterministic, every call).
  kan_prep_weights<<<(NT * OUT_DIM + 255) / 256, 256, 0, stream>>>(sw, bw, Bbf);

  dim3 grid((N_ROWS / BM) * (OUT_DIM / BN));      // 512 workgroups
  dim3 block(256);                                // 8 wave32 waves
  kan_wmma_bf16_kernel<<<grid, block, 0, stream>>>(x, Bbf, bb, out);
}
